// EncoderLayer_28819230556773
// MI455X (gfx1250) — compile-verified, matrix operand order checked by reading, not run on hardware
//
#include <hip/hip_runtime.h>
#include <hip/hip_bf16.h>
#include <math.h>

typedef unsigned short u16;
typedef unsigned int   u32;
typedef __attribute__((ext_vector_type(16))) __bf16 v16bf;
typedef __attribute__((ext_vector_type(8)))  float  v8f;
typedef __attribute__((ext_vector_type(16))) u16    u16x16;
typedef __attribute__((__vector_size__(4 * sizeof(int)))) int v4i;

#if defined(__has_builtin)
#if __has_builtin(__builtin_amdgcn_global_load_async_to_lds_b128)
#define USE_ASYNC_LDS 1
#endif
#endif
#ifndef USE_ASYNC_LDS
#define USE_ASYNC_LDS 0
#endif

// async copy helper pointer casts: builtin expects v4i* in AS1 (src) / AS3 (dst)
#define ASYNC_GLB(p) ((__attribute__((address_space(1))) v4i*)(p))
#define ASYNC_LDS(p) ((__attribute__((address_space(3))) v4i*)(p))

template <int N>
__device__ __forceinline__ void wait_asynccnt() {
#if defined(__has_builtin) && __has_builtin(__builtin_amdgcn_s_wait_asynccnt)
  __builtin_amdgcn_s_wait_asynccnt(N);
#else
  asm volatile("s_wait_asynccnt %0" ::"i"(N) : "memory");
#endif
}

__device__ __forceinline__ u16 f2bf(float f) {
  u32 u = __builtin_bit_cast(u32, f);
  u = (u + 0x7FFFu + ((u >> 16) & 1u)) >> 16;   // round-to-nearest-even
  return (u16)u;
}
__device__ __forceinline__ v8f vzero8() {
  v8f z;
#pragma unroll
  for (int i = 0; i < 8; ++i) z[i] = 0.0f;
  return z;
}
// K-offset of dword pair j for 16-bit A/B fragments (ISA 7.12.2: 16-bit A 16x32)
__device__ __forceinline__ int kpair(int j, int half) {
  return ((j >> 2) << 4) + ((j & 3) << 1) + (half << 3);
}

// ---------------------------------------------------------------------------
// Generic bf16 WMMA GEMM: C[M,N(+scatter)] = A[M,K] * Bt[N,K]^T + bias
// Optional: +=C (accumulate), residual add, ReLU, bf16 and/or f32 output,
// 2-segment output-column scatter (for the l_v combo).
// Tile: 128x64x32, 8 waves (wave32).
// Operand staging: double-buffered GLOBAL_LOAD_ASYNC_TO_LDS_B128 (ASYNCcnt),
// falling back to VGPR-staged copies if the builtin is unavailable.
// ---------------------------------------------------------------------------
#define BM 128
#define BN 64
#define BK 32

__global__ __launch_bounds__(256) void gemm_bf16_wmma(
    const u16* __restrict__ A, int lda,
    const u16* __restrict__ Bt,                  // [N][K] bf16
    const float* __restrict__ bias,              // [N] or null
    const float* __restrict__ residual, int ldr, // or null
    float* __restrict__ Cf, u16* __restrict__ Cb, int ldc,
    int M, int N, int K, int accumulate, int relu,
    int seg1, int off0, int off1)
{
  __shared__ u16 As[2][BM][BK];
  __shared__ u16 Bs[2][BN][BK];
  const int tid  = threadIdx.x;
  const int lane = tid & 31;
  const int wave = tid >> 5;
  const int bm = blockIdx.x * BM;
  const int bn = blockIdx.y * BN;
  const int wm = (wave & 3) * 32;
  const int wn = (wave >> 2) * 32;
  const int mrow = lane & 15;
  const int half = lane >> 4;

  v8f acc[2][2];
#pragma unroll
  for (int i = 0; i < 2; ++i)
#pragma unroll
    for (int j = 0; j < 2; ++j) acc[i][j] = vzero8();

  const int ar = tid >> 1, ah = (tid & 1) << 4;
  const int br = tid >> 2, bh = (tid & 3) << 3;
  const int T = K / BK;

  auto issue_tile = [&](int buf, int k0) {
    const u16* asrc = A + (size_t)(bm + ar) * lda + k0 + ah;
    const u16* bsrc = Bt + (size_t)(bn + br) * K + k0 + bh;
#if USE_ASYNC_LDS
    __builtin_amdgcn_global_load_async_to_lds_b128(
        ASYNC_GLB(asrc),     ASYNC_LDS(&As[buf][ar][ah]),     0, 0);
    __builtin_amdgcn_global_load_async_to_lds_b128(
        ASYNC_GLB(asrc + 8), ASYNC_LDS(&As[buf][ar][ah + 8]), 0, 0);
    __builtin_amdgcn_global_load_async_to_lds_b128(
        ASYNC_GLB(bsrc),     ASYNC_LDS(&Bs[buf][br][bh]),     0, 0);
#else
    *(uint4*)&As[buf][ar][ah]     = *(const uint4*)(asrc);
    *(uint4*)&As[buf][ar][ah + 8] = *(const uint4*)(asrc + 8);
    *(uint4*)&Bs[buf][br][bh]     = *(const uint4*)(bsrc);
    __builtin_prefetch(asrc + BK, 0, 1);
    __builtin_prefetch(bsrc + BK, 0, 1);
#endif
  };

  issue_tile(0, 0);
  for (int t = 0; t < T; ++t) {
    const int cb = t & 1;
    if (t + 1 < T) {
      issue_tile(cb ^ 1, (t + 1) * BK);   // prefetch next tile into other buffer
#if USE_ASYNC_LDS
      wait_asynccnt<3>();                 // the 3 just-issued may stay in flight
#endif
    } else {
#if USE_ASYNC_LDS
      wait_asynccnt<0>();
#endif
    }
    __syncthreads();                      // all waves' tile-t LDS writes visible

    v16bf afrag[2], bfrag[2];
#pragma unroll
    for (int i = 0; i < 2; ++i) {
      u16x16 tf;
      const int m = wm + i * 16 + mrow;
#pragma unroll
      for (int j = 0; j < 8; ++j) {
        const u32 p = *(const u32*)&As[cb][m][kpair(j, half)];
        tf[2 * j] = (u16)p; tf[2 * j + 1] = (u16)(p >> 16);
      }
      afrag[i] = __builtin_bit_cast(v16bf, tf);
    }
#pragma unroll
    for (int i = 0; i < 2; ++i) {
      u16x16 tf;
      const int n = wn + i * 16 + mrow;
#pragma unroll
      for (int j = 0; j < 8; ++j) {
        const u32 p = *(const u32*)&Bs[cb][n][kpair(j, half)];
        tf[2 * j] = (u16)p; tf[2 * j + 1] = (u16)(p >> 16);
      }
      bfrag[i] = __builtin_bit_cast(v16bf, tf);
    }
#pragma unroll
    for (int i = 0; i < 2; ++i)
#pragma unroll
      for (int j = 0; j < 2; ++j)
        acc[i][j] = __builtin_amdgcn_wmma_f32_16x16x32_bf16(
            false, afrag[i], false, bfrag[j], (short)0, acc[i][j], false, false);
    __syncthreads();                      // tile-t reads done; buffer reusable
  }

#pragma unroll
  for (int i = 0; i < 2; ++i)
#pragma unroll
    for (int j = 0; j < 2; ++j) {
      const int nl = bn + wn + j * 16 + mrow;
      const int gc = (nl < seg1) ? (nl + off0) : (nl + off1);
      const float bv = bias ? bias[nl] : 0.0f;
#pragma unroll
      for (int r = 0; r < 8; ++r) {
        const int m = bm + wm + i * 16 + half * 8 + r;
        float v = acc[i][j][r] + bv;
        if (residual) v += residual[(size_t)m * ldr + gc];
        const size_t o = (size_t)m * ldc + gc;
        if (accumulate) v += Cf[o];
        if (relu) v = v > 0.0f ? v : 0.0f;
        if (Cf) Cf[o] = v;
        if (Cb) Cb[o] = f2bf(v);
      }
    }
}

// ---------------------------------------------------------------------------
// Flash attention, one wave per (combo,b,h,16-row q tile). kv tiled by 32.
// Scores & ctx via WMMA; online softmax in registers; P staged in LDS;
// V tile staged via async LDS DMA, overlapped with the score phase.
// ---------------------------------------------------------------------------
template <int DK>
__global__ __launch_bounds__(32) void attn_flash(
    const u16* __restrict__ Qb, const u16* __restrict__ Kb, const u16* __restrict__ Vb,
    const unsigned char* __restrict__ maskp,   // [B][S][S] bool
    u16* __restrict__ Ctx, int S, int Dc, float scale)
{
  constexpr int KST = DK / 32;
  constexpr int DGR = DK / 16;
  __shared__ u16 Ps[16][32];
  __shared__ u16 Vs[32][DK];
  const int lane = threadIdx.x;
  const int mrow = lane & 15;
  const int half = lane >> 4;
  const int q0 = blockIdx.x * 16;
  const int b  = blockIdx.y >> 2;
  const int h  = blockIdx.y & 3;
  const size_t rowbase = (size_t)b * S;
  const int colQ = h * DK;

  v16bf qf[KST];
#pragma unroll
  for (int s = 0; s < KST; ++s) {
    u16x16 tf;
    const u16* qr = Qb + (rowbase + q0 + mrow) * Dc + colQ + s * 32;
#pragma unroll
    for (int j = 0; j < 8; ++j) {
      const u32 p = *(const u32*)(qr + kpair(j, half));
      tf[2 * j] = (u16)p; tf[2 * j + 1] = (u16)(p >> 16);
    }
    qf[s] = __builtin_bit_cast(v16bf, tf);
  }

  float rmax[8], rsum[8];
#pragma unroll
  for (int r = 0; r < 8; ++r) { rmax[r] = -3.0e38f; rsum[r] = 0.0f; }
  v8f ctx[DGR];
#pragma unroll
  for (int g = 0; g < DGR; ++g) ctx[g] = vzero8();

  for (int kv0 = 0; kv0 < S; kv0 += 32) {
    // stage V tile into LDS (async: overlaps with score WMMAs + softmax)
    const u16* vr = Vb + (rowbase + kv0 + lane) * Dc + colQ;
#if USE_ASYNC_LDS
#pragma unroll
    for (int d = 0; d < DK; d += 8)
      __builtin_amdgcn_global_load_async_to_lds_b128(
          ASYNC_GLB(vr + d), ASYNC_LDS(&Vs[lane][d]), 0, 0);
#else
#pragma unroll
    for (int d = 0; d < DK; d += 8)
      *(uint4*)&Vs[lane][d] = *(const uint4*)(vr + d);
#endif
    if (kv0 + 32 < S)
      __builtin_prefetch(Vb + (rowbase + kv0 + 32 + lane) * Dc + colQ, 0, 1);

    v8f sc0 = vzero8(), sc1 = vzero8();
#pragma unroll
    for (int s = 0; s < KST; ++s) {
      u16x16 t0, t1;
      const u16* kr0 = Kb + (rowbase + kv0 + mrow) * Dc + colQ + s * 32;
      const u16* kr1 = kr0 + (size_t)16 * Dc;
#pragma unroll
      for (int j = 0; j < 8; ++j) {
        const int kp = kpair(j, half);
        const u32 p0 = *(const u32*)(kr0 + kp);
        const u32 p1 = *(const u32*)(kr1 + kp);
        t0[2 * j] = (u16)p0; t0[2 * j + 1] = (u16)(p0 >> 16);
        t1[2 * j] = (u16)p1; t1[2 * j + 1] = (u16)(p1 >> 16);
      }
      sc0 = __builtin_amdgcn_wmma_f32_16x16x32_bf16(
          false, qf[s], false, __builtin_bit_cast(v16bf, t0), (short)0, sc0, false, false);
      sc1 = __builtin_amdgcn_wmma_f32_16x16x32_bf16(
          false, qf[s], false, __builtin_bit_cast(v16bf, t1), (short)0, sc1, false, false);
    }

#pragma unroll
    for (int r = 0; r < 8; ++r) {
      const int qrow = q0 + half * 8 + r;
      const unsigned char* mk = maskp + ((size_t)b * S + qrow) * S + kv0;
      float s0 = sc0[r] * scale; if (!mk[mrow])      s0 = -1.0e9f;
      float s1 = sc1[r] * scale; if (!mk[16 + mrow]) s1 = -1.0e9f;
      float mx = fmaxf(s0, s1);
#pragma unroll
      for (int o = 1; o < 16; o <<= 1) mx = fmaxf(mx, __shfl_xor(mx, o, 32));
      const float nm = fmaxf(rmax[r], mx);
      const float fr = __expf(rmax[r] - nm);
      const float e0 = __expf(s0 - nm);
      const float e1 = __expf(s1 - nm);
      float ssum = e0 + e1;
#pragma unroll
      for (int o = 1; o < 16; o <<= 1) ssum += __shfl_xor(ssum, o, 32);
      rsum[r] = rsum[r] * fr + ssum;
      rmax[r] = nm;
#pragma unroll
      for (int g = 0; g < DGR; ++g) ctx[g][r] *= fr;
      Ps[half * 8 + r][mrow]      = f2bf(e0);
      Ps[half * 8 + r][16 + mrow] = f2bf(e1);
    }
    __syncthreads();
#if USE_ASYNC_LDS
    wait_asynccnt<0>();                 // V tile resident before P @ V
#endif

    u16x16 tp;
#pragma unroll
    for (int j = 0; j < 8; ++j) {
      const u32 p = *(const u32*)&Ps[mrow][kpair(j, half)];
      tp[2 * j] = (u16)p; tp[2 * j + 1] = (u16)(p >> 16);
    }
    const v16bf pf = __builtin_bit_cast(v16bf, tp);
#pragma unroll
    for (int g = 0; g < DGR; ++g) {
      u16x16 tv;
      const int d = g * 16 + mrow;
#pragma unroll
      for (int j = 0; j < 8; ++j) {
        const int kp = kpair(j, half);
        tv[2 * j]     = Vs[kp][d];
        tv[2 * j + 1] = Vs[kp + 1][d];
      }
      ctx[g] = __builtin_amdgcn_wmma_f32_16x16x32_bf16(
          false, pf, false, __builtin_bit_cast(v16bf, tv), (short)0, ctx[g], false, false);
    }
    __syncthreads();
  }

#pragma unroll
  for (int g = 0; g < DGR; ++g) {
    const int d = colQ + g * 16 + mrow;
#pragma unroll
    for (int r = 0; r < 8; ++r) {
      const int m = q0 + half * 8 + r;
      Ctx[(rowbase + m) * Dc + d] = f2bf(ctx[g][r] / rsum[r]);
    }
  }
}

// ---------------------------------------------------------------------------
// LayerNorm (ddof=1) over 512 cols; optionally emits bf16 packed l_v rows and
// initializes Y = alpha_m * x + phi_m (the collapsed conv constant terms).
// One wave per row, 8 rows per block.
// ---------------------------------------------------------------------------
__global__ __launch_bounds__(256) void norm_rows(
    const float* __restrict__ X, int ldx,
    const float* __restrict__ gamma, const float* __restrict__ beta,
    u16* __restrict__ XN, u16* __restrict__ XLV,
    float* __restrict__ Y, const float* __restrict__ G, int M)
{
  const int row  = blockIdx.x * 8 + threadIdx.y;
  const int lane = threadIdx.x;
  if (row >= M) return;
  const float* x = X + (size_t)row * ldx;
  float vals[16];
  float s = 0.0f, s2 = 0.0f;
#pragma unroll
  for (int i = 0; i < 16; ++i) {
    const float v = x[lane * 16 + i];
    vals[i] = v; s += v; s2 += v * v;
  }
#pragma unroll
  for (int o = 1; o < 32; o <<= 1) {
    s  += __shfl_xor(s,  o, 32);
    s2 += __shfl_xor(s2, o, 32);
  }
  const float mu  = s * (1.0f / 512.0f);
  const float var = fmaxf((s2 - s * mu) * (1.0f / 511.0f), 0.0f);  // ddof=1
  const float inv = 1.0f / (sqrtf(var) + 1e-6f);
#pragma unroll
  for (int i = 0; i < 16; ++i) {
    const int c = lane * 16 + i;
    const float xn = gamma[c] * (vals[i] - mu) * inv + beta[c];
    XN[(size_t)row * 512 + c] = f2bf(xn);
    if (XLV) {
      if (c < 256)       XLV[(size_t)row * 384 + c]       = f2bf(xn);
      else if (c >= 384) XLV[(size_t)row * 384 + c - 128] = f2bf(xn);
    }
    if (Y) {
      const int mm = (c < 256) ? 0 : ((c < 384) ? 1 : 2);
      Y[(size_t)row * 512 + c] = G[48 + mm] * vals[i] + G[51 + mm];
    }
  }
}

// fp32 [K,N] -> bf16 [N,K]
__global__ __launch_bounds__(256) void wt_transpose(
    const float* __restrict__ W, u16* __restrict__ Wt, int K, int N)
{
  const int idx = blockIdx.x * 256 + threadIdx.x;
  if (idx >= K * N) return;
  const int n = idx / K, k = idx % K;
  Wt[idx] = f2bf(W[(size_t)k * N + n]);
}

// Collapse conv stack per head: e[h][4] = W0*W1*W2, f[h] = (b0*W1 + b1)*W2 + b2
__global__ void conv_compose(
    const float* __restrict__ W0, const float* __restrict__ b0,
    const float* __restrict__ W1, const float* __restrict__ b1,
    const float* __restrict__ W2, const float* __restrict__ b2,
    float* __restrict__ ef)
{
  const int h = threadIdx.x;   // 4 threads
  float t[8];
  for (int j = 0; j < 8; ++j) {
    float a = 0.0f;
    for (int o = 0; o < 4; ++o) a += W1[h * 32 + j * 4 + o] * W2[h * 4 + o];
    t[j] = a;
  }
  for (int i = 0; i < 4; ++i) {
    float a = 0.0f;
    for (int j = 0; j < 8; ++j) a += W0[h * 32 + i * 8 + j] * t[j];
    ef[h * 4 + i] = a;
  }
  float f = 0.0f;
  for (int j = 0; j < 8; ++j) f += b0[h * 8 + j] * t[j];
  for (int o = 0; o < 4; ++o) f += b1[h * 4 + o] * W2[h * 4 + o];
  f += b2[h];
  ef[16 + h] = f;
}

// Suffix sums over heads (cumsum collapse): g[m][i][h'] = sum_{h>=h'} e[h][i];
// alpha[m] = sum e; phi[m] = sum f.
__global__ void conv_finalize(const float* __restrict__ EF, float* __restrict__ G)
{
  const int m = threadIdx.x;
  if (m >= 3) return;
  const float* ef = EF + m * 20;
  for (int i = 0; i < 4; ++i)
    for (int hp = 0; hp < 4; ++hp) {
      float s = 0.0f;
      for (int h = hp; h < 4; ++h) s += ef[h * 4 + i];
      G[m * 16 + i * 4 + hp] = s;
    }
  float alpha = 0.0f, phi = 0.0f;
  for (int h = 0; h < 4; ++h) {
    for (int i = 0; i < 4; ++i) alpha += ef[h * 4 + i];
    phi += ef[16 + h];
  }
  G[48 + m] = alpha;
  G[51 + m] = phi;
}

// Scale Wo[h] columns by the per-(modality,appearance,head) suffix weight g,
// transpose to bf16 [Dc][Dc]; also emit weighted bias vector.
__global__ __launch_bounds__(512) void wo_prep(
    const float* __restrict__ Wo, const float* __restrict__ bo,
    const float* __restrict__ G, u16* __restrict__ Wt, float* __restrict__ ybias,
    int Dc, int dk, int s0, int s1, int g0, int g1, int g2)
{
  const int n = blockIdx.x;                          // output column (local)
  const int gsel = (n < s0) ? g0 : ((n < s0 + s1) ? g1 : g2);
  const float* g = G + gsel * 4;
  const int t = threadIdx.x;
  if (t < Dc) {
    const int h = t / dk;
    Wt[(size_t)n * Dc + t] = f2bf(Wo[(size_t)t * Dc + n] * g[h]);
  }
  if (t == 0) {
    float s = 0.0f;
    for (int h = 0; h < 4; ++h) s += g[h] * bo[h * Dc + n];
    ybias[n] = s;
  }
}

// ---------------------------------------------------------------------------
extern "C" void kernel_launch(void* const* d_in, const int* in_sizes, int n_in,
                              void* d_out, int out_size, void* d_ws, size_t ws_size,
                              hipStream_t stream)
{
  (void)in_sizes; (void)n_in; (void)out_size; (void)ws_size;
  const int B = 32, S = 512, M = B * S;

  int cur = 0;
  const float* x_concat = (const float*)d_in[cur++];
  const unsigned char* masks = (const unsigned char*)d_in[cur++];
  const float *Wq[7], *bq[7], *Wk[7], *bk[7], *Wv[7], *bv[7], *Wo[7], *bo[7];
  for (int c = 0; c < 7; ++c) {
    Wq[c] = (const float*)d_in[cur++]; bq[c] = (const float*)d_in[cur++];
    Wk[c] = (const float*)d_in[cur++]; bk[c] = (const float*)d_in[cur++];
    Wv[c] = (const float*)d_in[cur++]; bv[c] = (const float*)d_in[cur++];
    Wo[c] = (const float*)d_in[cur++]; bo[c] = (const float*)d_in[cur++];
  }
  const float* cw[3][6];
  for (int m = 0; m < 3; ++m)
    for (int j = 0; j < 6; ++j) cw[m][j] = (const float*)d_in[cur++];
  const float* n1g = (const float*)d_in[cur++];
  const float* n1b = (const float*)d_in[cur++];
  const float* n2g = (const float*)d_in[cur++];
  const float* n2b = (const float*)d_in[cur++];
  const float* fW1 = (const float*)d_in[cur++];
  const float* fb1 = (const float*)d_in[cur++];
  const float* fW2 = (const float*)d_in[cur++];
  const float* fb2 = (const float*)d_in[cur++];

  char* wp = (char*)d_ws;
  auto alloc = [&](size_t bytes) {
    void* p = (void*)wp;
    wp += (bytes + 255) & ~(size_t)255;
    return p;
  };
  float* G  = (float*)alloc(256 * 4);
  u16* XN   = (u16*)alloc((size_t)M * 512 * 2);
  u16* XLV  = (u16*)alloc((size_t)M * 384 * 2);
  u16* Qb   = (u16*)alloc((size_t)M * 512 * 2);
  u16* Kb   = (u16*)alloc((size_t)M * 512 * 2);
  u16* Vb   = (u16*)alloc((size_t)M * 512 * 2);
  u16* CTX  = (u16*)alloc((size_t)M * 512 * 2);
  float* Y  = (float*)alloc((size_t)M * 512 * 4);
  u16* X2N  = (u16*)alloc((size_t)M * 512 * 2);
  u16* FFH  = (u16*)alloc((size_t)M * 1024 * 2);
  u16* WT   = (u16*)alloc((size_t)1024 * 512 * 2);
  float* YB = (float*)alloc(512 * 4);

  // 1) collapse the conv stacks into per-head affine weights + suffix sums
  for (int m = 0; m < 3; ++m)
    conv_compose<<<1, 4, 0, stream>>>(cw[m][0], cw[m][1], cw[m][2], cw[m][3],
                                      cw[m][4], cw[m][5], G + 64 + m * 20);
  conv_finalize<<<1, 4, 0, stream>>>(G + 64, G);

  // 2) norm1 -> bf16 (incl. packed l_v rows), Y init = alpha*x + phi
  norm_rows<<<dim3(M / 8), dim3(32, 8), 0, stream>>>(x_concat, 512, n1g, n1b,
                                                     XN, XLV, Y, G, M);

  struct ComboMeta { int Dc, inMode, colOff, seg1, off0, off1, s0, s1, g0, g1, g2; };
  static const ComboMeta cm[7] = {
    // Dc  in colO seg1 off0 off1  s0   s1  g0  g1  g2
    {256, 0,   0, 256,   0,   0, 256,   0,  0,  0,  0},  // l
    {128, 0, 256, 128, 256, 256, 128,   0,  4,  4,  4},  // a
    {128, 0, 384, 128, 384, 384, 128,   0,  8,  8,  8},  // v
    {384, 0,   0, 384,   0,   0, 256, 128,  1,  5,  5},  // l_a
    {384, 1,   0, 256,   0, 128, 256, 128,  2,  9,  9},  // l_v (col scatter)
    {256, 0, 256, 256, 256, 256, 128, 128,  6, 10, 10},  // a_v
    {512, 0,   0, 512,   0,   0, 256, 128,  3,  7, 11},  // l_a_v
  };

  for (int ci = 0; ci < 7; ++ci) {
    const ComboMeta& c = cm[ci];
    const int Dc = c.Dc, dk = Dc / 4;
    const u16* Ain = c.inMode ? XLV : (XN + c.colOff);
    const int lda = c.inMode ? 384 : 512;
    const dim3 ggrid(M / 128, Dc / 64);
    const int wtBlocks = (Dc * Dc + 255) / 256;

    wt_transpose<<<wtBlocks, 256, 0, stream>>>(Wq[ci], WT, Dc, Dc);
    gemm_bf16_wmma<<<ggrid, 256, 0, stream>>>(Ain, lda, WT, bq[ci], nullptr, 0,
                                              nullptr, Qb, Dc, M, Dc, Dc, 0, 0, Dc, 0, 0);
    wt_transpose<<<wtBlocks, 256, 0, stream>>>(Wk[ci], WT, Dc, Dc);
    gemm_bf16_wmma<<<ggrid, 256, 0, stream>>>(Ain, lda, WT, bk[ci], nullptr, 0,
                                              nullptr, Kb, Dc, M, Dc, Dc, 0, 0, Dc, 0, 0);
    wt_transpose<<<wtBlocks, 256, 0, stream>>>(Wv[ci], WT, Dc, Dc);
    gemm_bf16_wmma<<<ggrid, 256, 0, stream>>>(Ain, lda, WT, bv[ci], nullptr, 0,
                                              nullptr, Vb, Dc, M, Dc, Dc, 0, 0, Dc, 0, 0);

    const float scale = 1.0f / sqrtf((float)dk);
    const unsigned char* mp = masks + (size_t)ci * B * S * S;
    const dim3 agrid(S / 16, B * 4);
    switch (dk) {
      case 32:  attn_flash<32> <<<agrid, 32, 0, stream>>>(Qb, Kb, Vb, mp, CTX, S, Dc, scale); break;
      case 64:  attn_flash<64> <<<agrid, 32, 0, stream>>>(Qb, Kb, Vb, mp, CTX, S, Dc, scale); break;
      case 96:  attn_flash<96> <<<agrid, 32, 0, stream>>>(Qb, Kb, Vb, mp, CTX, S, Dc, scale); break;
      default:  attn_flash<128><<<agrid, 32, 0, stream>>>(Qb, Kb, Vb, mp, CTX, S, Dc, scale); break;
    }

    // collapsed conv/cumsum/head-sum: Y += CTX @ (g-scaled Wo) + weighted bo
    wo_prep<<<dim3(Dc), 512, 0, stream>>>(Wo[ci], bo[ci], G, WT, YB, Dc, dk,
                                          c.s0, c.s1, c.g0, c.g1, c.g2);
    gemm_bf16_wmma<<<ggrid, 256, 0, stream>>>(CTX, Dc, WT, YB, nullptr, 0,
                                              Y, nullptr, 512, M, Dc, Dc, 1, 0,
                                              c.seg1, c.off0, c.off1);
  }

  // 3) norm2 -> bf16
  norm_rows<<<dim3(M / 8), dim3(32, 8), 0, stream>>>(Y, 512, n2g, n2b,
                                                     X2N, nullptr, nullptr, nullptr, M);

  // 4) FFN: relu(X2N@W1+b1)@W2+b2 + x_concat -> out (fp32)
  wt_transpose<<<(512 * 1024 + 255) / 256, 256, 0, stream>>>(fW1, WT, 512, 1024);
  gemm_bf16_wmma<<<dim3(M / 128, 1024 / 64), 256, 0, stream>>>(
      X2N, 512, WT, fb1, nullptr, 0, nullptr, FFH, 1024, M, 1024, 512, 0, 1, 1024, 0, 0);
  wt_transpose<<<(1024 * 512 + 255) / 256, 256, 0, stream>>>(fW2, WT, 1024, 512);
  gemm_bf16_wmma<<<dim3(M / 128, 512 / 64), 256, 0, stream>>>(
      FFH, 1024, WT, fb2, x_concat, 512, (float*)d_out, nullptr, 512,
      M, 512, 1024, 0, 0, 512, 0, 0);
}
